// GCN_18691697672407
// MI455X (gfx1250) — matrix operand order repaired and checked by reading, not compiled, not run on hardware
//
#include <hip/hip_runtime.h>
#include <hip/hip_bf16.h>
#include <stdint.h>

// ---------------------------------------------------------------------------
// GCN forward:  3x [ bf16-WMMA GEMM (LDS async-staged B) -> SpMM -> bias ]
// ---------------------------------------------------------------------------

#define NNODES 10000
#define NEDGES 320000
#define MT     625          // NNODES / 16 row tiles

typedef __attribute__((ext_vector_type(16))) __bf16 v16bf;
typedef __attribute__((ext_vector_type(8)))  float  v8f;
typedef __attribute__((ext_vector_type(4)))  int    v4i;

union ABReg {
    uint4 u[2];     // 2 x b128 loads
    v16bf v;        // WMMA operand view
};

// ---------------- CDNA5 async global->LDS staging --------------------------
#if __has_builtin(__builtin_amdgcn_global_load_async_to_lds_b128)
#define ASYNC_LDS_BUILTIN 1
#else
#define ASYNC_LDS_BUILTIN 0
#endif

typedef __attribute__((address_space(1))) v4i g_v4i;   // global b128 chunk
typedef __attribute__((address_space(3))) v4i l_v4i;   // LDS b128 chunk

__device__ __forceinline__ void stage_b128(const uint32_t* gsrc, uint32_t* ldst) {
#if ASYNC_LDS_BUILTIN
    __builtin_amdgcn_global_load_async_to_lds_b128(
        (g_v4i*)(uintptr_t)gsrc,
        (l_v4i*)(uint32_t)(uintptr_t)ldst, 0, 0);
#else
    // inline-asm fallback: VDST = LDS byte address, VADDR = 64-bit global addr
    uint32_t lo = (uint32_t)(uintptr_t)ldst;      // low 32 bits of generic = LDS addr
    uint64_t ga = (uint64_t)(uintptr_t)gsrc;
    asm volatile("global_load_async_to_lds_b128 %0, %1, off"
                 :: "v"(lo), "v"(ga) : "memory");
#endif
}

__device__ __forceinline__ void wait_async0() {
#if __has_builtin(__builtin_amdgcn_s_wait_asynccnt)
    __builtin_amdgcn_s_wait_asynccnt(0);
#else
    asm volatile("s_wait_asynccnt 0" ::: "memory");
#endif
}

__device__ __forceinline__ void load2(ABReg& r, const uint32_t* p) {
    r.u[0] = *(const uint4*)(p);
    r.u[1] = *(const uint4*)(p + 4);
}

// float -> bf16 (round to nearest even), raw bits
__device__ __forceinline__ uint32_t f2bf_raw(float f) {
    uint32_t u = __float_as_uint(f);
    return (u + 0x7FFFu + ((u >> 16) & 1u)) >> 16;
}
__device__ __forceinline__ uint32_t pack2bf(float lo, float hi) {
    return f2bf_raw(lo) | (f2bf_raw(hi) << 16);
}

// ---------------------------------------------------------------------------
// Pack activations (f32, ld=512) into swizzled bf16 A-tiles (16x32 per tile).
// tile = mt*Ktiles + kt ; in-tile dword = lane*8 + j (lane-contiguous 32B).
// A layout (ISA 7.12.2): lane = half*16+m ; j<4: K=half*8+2j ; j>=4: K=16+half*8+2(j-4)
// mode 0: v = src[m][k]            (layer-0 x, K=512)
// mode 1: v = k<512 ? relu(S[m][k]) : S[m][k-512]   (concat layers, K=1024)
// ---------------------------------------------------------------------------
__global__ void __launch_bounds__(256)
pack_act_kernel(const float* __restrict__ src, uint32_t* __restrict__ dst,
                int Ktiles, int mode, int total) {
    int tid = blockIdx.x * 256 + threadIdx.x;
    if (tid >= total) return;
    int j    = tid & 7;
    int lane = (tid >> 3) & 31;
    int tile = tid >> 8;
    int kt   = tile % Ktiles;
    int mt   = tile / Ktiles;
    int half = lane >> 4;
    int m    = mt * 16 + (lane & 15);
    int k    = kt * 32 + ((j < 4) ? (half * 8 + 2 * j) : (16 + half * 8 + 2 * (j - 4)));

    float v0, v1;
    if (mode == 0) {
        v0 = src[(size_t)m * 512 + k];
        v1 = src[(size_t)m * 512 + k + 1];
    } else {
        float s0 = src[(size_t)m * 512 + (k & 511)];
        float s1 = src[(size_t)m * 512 + ((k + 1) & 511)];
        v0 = (k < 512) ? fmaxf(s0, 0.f) : s0;
        v1 = (k < 512) ? fmaxf(s1, 0.f) : s1;
    }
    dst[(size_t)tile * 256 + lane * 8 + j] = pack2bf(v0, v1);
}

// ---------------------------------------------------------------------------
// Pack weights (f32, KxN row-major) into swizzled bf16 B-tiles (32x16).
// tile = nt*Ktiles + kt ; B layout: lane = half*16+n ; j<4: K=half*8+2j ; j>=4: K=16+half*8+2(j-4)
// ---------------------------------------------------------------------------
__global__ void __launch_bounds__(256)
pack_w_kernel(const float* __restrict__ W, uint32_t* __restrict__ dst,
              int Ktiles, int N, int total) {
    int tid = blockIdx.x * 256 + threadIdx.x;
    if (tid >= total) return;
    int j    = tid & 7;
    int lane = (tid >> 3) & 31;
    int tile = tid >> 8;
    int kt   = tile % Ktiles;
    int nt   = tile / Ktiles;
    int k    = kt * 32 + ((j < 4) ? ((lane >> 4) * 8 + 2 * j) : (16 + (lane >> 4) * 8 + 2 * (j - 4)));
    int col  = nt * 16 + (lane & 15);
    dst[(size_t)tile * 256 + lane * 8 + j] =
        pack2bf(W[(size_t)k * N + col], W[(size_t)(k + 1) * N + col]);
}

// ---------------------------------------------------------------------------
// GEMM: Y[10000 x N] = A @ B (bf16 WMMA, f32 accumulate).
// Block = 8 waves = 128 rows, all waves share one 64-col N-group (blockIdx.y).
// Per K-step: block async-stages 4KB of B into LDS (double-buffered, ASYNCcnt),
// each wave reads it back with ds_load_b128 and runs 4x v_wmma_16x16x32_bf16.
// A is register double-buffered from global (immediate-offset b128 loads).
// ---------------------------------------------------------------------------
template<int KT>
__global__ void __launch_bounds__(256)
gemm_wmma_kernel(const uint32_t* __restrict__ A, const uint32_t* __restrict__ B,
                 float* __restrict__ Y, int Nt /* N/16 */) {
    __shared__ uint32_t sB[2 * 1024];            // 2 x (4 tiles x 256 dwords) = 8KB

    const int tid  = threadIdx.x;
    const int lane = tid & 31;
    const int wave = tid >> 5;
    int mt = blockIdx.x * 8 + wave;
    if (mt > MT - 1) mt = MT - 1;                // clamp (no early-exit: keep barriers/EXEC)
    const int ng = blockIdx.y;
    const int N  = Nt * 16;

    // A: per-wave, register double-buffered
    const uint32_t* ap = A + (size_t)mt * (KT * 256) + lane * 8;

    // B staging: thread tid copies one b128; tile st, chunk sq
    const int st = tid >> 6;                     // 0..3
    const int sq = tid & 63;                     // 0..63
    const uint32_t* gsrc = B + (size_t)(ng * 4 + st) * (KT * 256) + sq * 4;
    const int ldsoff = st * 256 + sq * 4;

    ABReg a_cur, a_nxt;
    v8f acc[4];
    const v8f vzero = {0.f, 0.f, 0.f, 0.f, 0.f, 0.f, 0.f, 0.f};
#pragma unroll
    for (int t = 0; t < 4; ++t) acc[t] = vzero;

    // prologue: stage kt=0 and load A(0)
    stage_b128(gsrc, &sB[ldsoff]);
    load2(a_cur, ap);
    wait_async0();
    __syncthreads();

#pragma unroll
    for (int kt = 0; kt < KT; ++kt) {
        const int buf = (kt & 1) << 10;

        // read this step's B panel from LDS (ds_load_b128 x8 per wave)
        ABReg b[4];
#pragma unroll
        for (int t = 0; t < 4; ++t) {
            const uint32_t* lp = &sB[buf + t * 256 + lane * 8];
            b[t].u[0] = *(const uint4*)(lp);
            b[t].u[1] = *(const uint4*)(lp + 4);
        }

        // prefetch next step: async B -> other LDS buffer, A -> registers
        if (kt + 1 < KT) {
            stage_b128(gsrc + (kt + 1) * 256, &sB[(buf ^ 1024) + ldsoff]);
            load2(a_nxt, ap + (kt + 1) * 256);
        }

#pragma unroll
        for (int t = 0; t < 4; ++t)
            acc[t] = __builtin_amdgcn_wmma_f32_16x16x32_bf16(
                false, a_cur.v, false, b[t].v, (short)0, acc[t], false, false);

        if (kt + 1 < KT) {
            a_cur = a_nxt;
            wait_async0();                        // my stage(kt+1) landed
        }
        __syncthreads();                          // whole block's panel ready / reads done
    }

    // C/D layout: VGPR r, lanes 0-15 -> M=r, N=lane ; lanes 16-31 -> M=r+8
    const int mrow = mt * 16 + (lane >> 4) * 8;
    const int ncol = ng * 64 + (lane & 15);
#pragma unroll
    for (int t = 0; t < 4; ++t) {
        size_t base = (size_t)mrow * N + ncol + t * 16;
#pragma unroll
        for (int r = 0; r < 8; ++r)
            Y[base + (size_t)r * N] = acc[t][r];
    }
}

// ---------------------------------------------------------------------------
__global__ void __launch_bounds__(256)
init_bias_kernel(float* __restrict__ out, const float* __restrict__ b,
                 int dmask, int total) {
    int tid = blockIdx.x * 256 + threadIdx.x;
    if (tid >= total) return;
    out[tid] = b[tid & dmask];
}

// ---------------------------------------------------------------------------
// SpMM: out[row] += val * Y[col] per edge; one block per edge, float4 chunks,
// f32 global atomics (operands are L2-resident: Y <= 20MB, out <= 20MB).
// ---------------------------------------------------------------------------
__global__ void spmm_kernel(const int* __restrict__ er, const int* __restrict__ ec,
                            const float* __restrict__ ev, const float* __restrict__ Yin,
                            float* __restrict__ out, int d) {
    const int e   = blockIdx.x;
    const int row = er[e];
    const int col = ec[e];
    const float v = ev[e];
    const int f   = threadIdx.x << 2;            // blockDim.x == d/4
    const float4 y = *(const float4*)(Yin + (size_t)col * d + f);
    float* o = out + (size_t)row * d + f;
    atomicAdd(o + 0, v * y.x);
    atomicAdd(o + 1, v * y.y);
    atomicAdd(o + 2, v * y.z);
    atomicAdd(o + 3, v * y.w);
}

// ---------------------------------------------------------------------------

static inline int cdiv(int a, int b) { return (a + b - 1) / b; }

extern "C" void kernel_launch(void* const* d_in, const int* in_sizes, int n_in,
                              void* d_out, int out_size, void* d_ws, size_t ws_size,
                              hipStream_t stream) {
    (void)in_sizes; (void)n_in; (void)out_size; (void)ws_size;
    const float* x  = (const float*)d_in[0];
    const int*   er = (const int*)  d_in[1];
    const int*   ec = (const int*)  d_in[2];
    const float* ev = (const float*)d_in[3];
    const float* W0 = (const float*)d_in[4];
    const float* W1 = (const float*)d_in[5];
    const float* W2 = (const float*)d_in[6];
    const float* b0 = (const float*)d_in[7];
    const float* b1 = (const float*)d_in[8];
    const float* b2 = (const float*)d_in[9];
    float* out = (float*)d_out;

    // workspace partition (~62.5 MB)
    float*    Y    = (float*)d_ws;                          // 10000*512 f32
    float*    S    = Y + (size_t)NNODES * 512;              // 10000*512 f32
    uint32_t* Xswz = (uint32_t*)(S + (size_t)NNODES * 512); // 10000*1024 bf16
    uint32_t* Wswz = Xswz + (size_t)NNODES * 1024 / 2;      // up to 1024*512 bf16

    const int gx = cdiv(MT, 8);                             // 79 row-blocks

    // ----- layer 0: x @ W0 (K=512, N=512) -----
    {
        int total = MT * 16 * 256;
        pack_act_kernel<<<cdiv(total, 256), 256, 0, stream>>>(x, Xswz, 16, 0, total);
        int wt = 32 * 16 * 256;
        pack_w_kernel<<<cdiv(wt, 256), 256, 0, stream>>>(W0, Wswz, 16, 512, wt);
        gemm_wmma_kernel<16><<<dim3(gx, 8), 256, 0, stream>>>(Xswz, Wswz, Y, 32);
        int bt = NNODES * 512;
        init_bias_kernel<<<cdiv(bt, 256), 256, 0, stream>>>(S, b0, 511, bt);
        spmm_kernel<<<NEDGES, 128, 0, stream>>>(er, ec, ev, Y, S, 512);
    }
    // ----- layer 1: concat(relu(S), S) @ W1 (K=1024, N=512) -----
    {
        int total = MT * 32 * 256;
        pack_act_kernel<<<cdiv(total, 256), 256, 0, stream>>>(S, Xswz, 32, 1, total);
        int wt = 32 * 32 * 256;
        pack_w_kernel<<<cdiv(wt, 256), 256, 0, stream>>>(W1, Wswz, 32, 512, wt);
        gemm_wmma_kernel<32><<<dim3(gx, 8), 256, 0, stream>>>(Xswz, Wswz, Y, 32);
        int bt = NNODES * 512;
        init_bias_kernel<<<cdiv(bt, 256), 256, 0, stream>>>(S, b1, 511, bt);
        spmm_kernel<<<NEDGES, 128, 0, stream>>>(er, ec, ev, Y, S, 512);
    }
    // ----- layer 2: concat(relu(S), S) @ W2 (K=1024, N=256) -> d_out -----
    {
        int total = MT * 32 * 256;
        pack_act_kernel<<<cdiv(total, 256), 256, 0, stream>>>(S, Xswz, 32, 1, total);
        int wt = 16 * 32 * 256;
        pack_w_kernel<<<cdiv(wt, 256), 256, 0, stream>>>(W2, Wswz, 32, 256, wt);
        gemm_wmma_kernel<32><<<dim3(gx, 4), 256, 0, stream>>>(Xswz, Wswz, Y, 16);
        int bt = NNODES * 256;
        init_bias_kernel<<<cdiv(bt, 256), 256, 0, stream>>>(out, b2, 255, bt);
        spmm_kernel<<<NEDGES, 64, 0, stream>>>(er, ec, ev, Y, out, 256);
    }
}
